// COIL_15118284882567
// MI455X (gfx1250) — matrix operand order repaired and checked by reading, not compiled, not run on hardware
//
#include <hip/hip_runtime.h>

typedef __attribute__((ext_vector_type(16))) _Float16 v16h;
typedef __attribute__((ext_vector_type(8)))  _Float16 v8h;
typedef __attribute__((ext_vector_type(4)))  _Float16 v4h;
typedef __attribute__((ext_vector_type(8)))  float    v8f;

#define BQ   64
#define LQ   32
#define BD   128
#define LD   192
#define TOKD 32
#define CLSD 768
#define QPB  8     // queries (waves) per block

// Concatenate two aligned 16-byte LDS chunks into one WMMA fragment (pure
// register concatenation, no ALU).
__device__ __forceinline__ v16h load_frag(const _Float16* p_lo, const _Float16* p_hi) {
    v8h lo = *(const v8h*)p_lo;
    v8h hi = *(const v8h*)p_hi;
    return __builtin_shufflevector(lo, hi, 0, 1, 2, 3, 4, 5, 6, 7,
                                           8, 9, 10, 11, 12, 13, 14, 15);
}

// Block = (doc d, group of 8 queries); 8 wave32s per block. The 24 KB doc
// tile is staged+converted to f16 ONCE per block by all 256 threads, then
// each wave runs the dense 16x16x32 f16 WMMA GEMM for its own query against
// the shared LDS tile. Masked-max is applied directly in the documented C/D
// register layout:
//   D VGPR r: lanes 0-15 -> (M=r, N=lane), lanes 16-31 -> (M=r+8, N=lane-16)
__global__ __launch_bounds__(32 * QPB) void coil_score_kernel(
    const float* __restrict__ qtok,   // [BQ, LQ, TOKD]
    const float* __restrict__ dtok,   // [BD, LD, TOKD]
    const float* __restrict__ qcls,   // [BQ, CLSD]
    const float* __restrict__ dcls,   // [BD, CLSD]
    const int*   __restrict__ qid,    // [BQ, LQ]
    const int*   __restrict__ did,    // [BD, LD]
    const int*   __restrict__ qmask,  // [BQ, LQ]
    float*       __restrict__ out)    // [BQ, BD]
{
    __shared__ alignas(16) _Float16 sDh[LD * TOKD];        // 12 KB doc tile, f16
    __shared__ alignas(16) _Float16 sQh[QPB][LQ * TOKD];   // 16 KB query tiles, f16
    __shared__ int   sDid[LD];
    __shared__ int   sQid[QPB][LQ];
    __shared__ float sQm[QPB][LQ];

    const int d    = blockIdx.x;                 // doc index
    const int wave = threadIdx.x >> 5;           // 0..7
    const int lane = threadIdx.x & 31;           // 0..31 (wave32)
    const int q    = blockIdx.y * QPB + wave;    // query index
    const int tid  = threadIdx.x;

    // ---- doc tile: staged by the whole block, f32 -> f16 converted once ----
    {
        const float4* dsrc = (const float4*)(dtok + (size_t)d * LD * TOKD);
        #pragma unroll
        for (int i = tid; i < LD * TOKD / 4; i += 32 * QPB) {
            float4 x = dsrc[i];
            v4h h = { (_Float16)x.x, (_Float16)x.y, (_Float16)x.z, (_Float16)x.w };
            *(v4h*)(sDh + i * 4) = h;            // aligned b64 ds store
        }
        if (tid < LD) sDid[tid] = did[d * LD + tid];
    }

    // ---- query tile: staged per wave ----
    {
        const float4* qsrc = (const float4*)(qtok + (size_t)q * LQ * TOKD);
        #pragma unroll
        for (int i = lane; i < LQ * TOKD / 4; i += 32) {
            float4 x = qsrc[i];
            v4h h = { (_Float16)x.x, (_Float16)x.y, (_Float16)x.z, (_Float16)x.w };
            *(v4h*)(&sQh[wave][i * 4]) = h;
        }
        sQid[wave][lane] = qid[q * LQ + lane];

        // qmask: zero the [SEP] position (last attended token); CLS (i==0) dropped.
        int m   = qmask[q * LQ + lane];
        int tot = m;
        for (int off = 1; off < 32; off <<= 1) tot += __shfl_xor(tot, off);
        int sep = tot - 1;
        sQm[wave][lane] = (lane >= 1 && m != 0 && lane != sep) ? 1.0f : 0.0f;
    }

    __syncthreads();

    const int hi  = (lane >= 16) ? 1 : 0;
    const int l16 = lane & 15;

    // ---- A fragments (16-bit A 16x32, ISA 7.12.2) ----
    // lane row M = lane%16; halves 0..15 = K 16*hi..16*hi+15 (this code's contiguous order, same in B).
    v16h afr[2];
    #pragma unroll
    for (int t = 0; t < 2; ++t) {
        const _Float16* row = &sQh[wave][(t * 16 + l16) * TOKD];
        afr[t] = load_frag(row + 16 * hi, row + 16 * hi + 8);
    }

    int myqid[2][8];   // query token ids for this lane's 16 M rows
    #pragma unroll
    for (int t = 0; t < 2; ++t)
        #pragma unroll
        for (int r = 0; r < 8; ++r) myqid[t][r] = sQid[wave][t * 16 + r + 8 * hi];

    // ---- jt = 0 peeled: initialize running max from first masked tile ----
    float runmax[2][8];
    {
        const _Float16* brow = sDh + l16 * TOKD + 16 * hi;
        v16h bfr = load_frag(brow, brow + 8);
        const int dtokid = sDid[l16];
        #pragma unroll
        for (int t = 0; t < 2; ++t) {
            v8f c = {};
            v8f s = __builtin_amdgcn_wmma_f32_16x16x32_f16(
                false, afr[t], false, bfr, (short)0, c, false, false);
            #pragma unroll
            for (int r = 0; r < 8; ++r)
                runmax[t][r] = (myqid[t][r] == dtokid) ? s[r] : 0.0f;
        }
    }

    // ---- remaining 11 doc-position tiles of 16 ----
    #pragma unroll
    for (int jt = 1; jt < LD / 16; ++jt) {
        // B fragment (16-bit B 32x16): N = lane%16, halves 0..15 = K 16*hi..+15
        const _Float16* brow = sDh + (jt * 16 + l16) * TOKD + 16 * hi;
        v16h bfr = load_frag(brow, brow + 8);
        const int dtokid = sDid[jt * 16 + l16];   // token id of this lane's N column

        #pragma unroll
        for (int t = 0; t < 2; ++t) {
            v8f c = {};
            v8f s = __builtin_amdgcn_wmma_f32_16x16x32_f16(
                false, afr[t], false, bfr, (short)0, c, false, false);
            #pragma unroll
            for (int r = 0; r < 8; ++r) {
                float v = (myqid[t][r] == dtokid) ? s[r] : 0.0f;  // where(match, s, 0)
                runmax[t][r] = fmaxf(runmax[t][r], v);
            }
        }
    }

    // ---- max over the 16 N-lanes per group, then masked sum over query tokens ----
    float sumtok = 0.0f;
    #pragma unroll
    for (int t = 0; t < 2; ++t) {
        float part = 0.0f;
        #pragma unroll
        for (int r = 0; r < 8; ++r) {
            float v = runmax[t][r];
            v = fmaxf(v, __shfl_xor(v, 1));
            v = fmaxf(v, __shfl_xor(v, 2));
            v = fmaxf(v, __shfl_xor(v, 4));
            v = fmaxf(v, __shfl_xor(v, 8));                   // max over doc columns
            part += v * sQm[wave][t * 16 + r + 8 * hi];
        }
        part += __shfl_xor(part, 16);                          // combine M halves
        sumtok += part;
    }

    // ---- CLS score: dot over 768, 24 elems/lane + wave32 reduction ----
    float acc = 0.0f;
    {
        const float* qc = qcls + (size_t)q * CLSD;
        const float* dc = dcls + (size_t)d * CLSD;
        for (int k = lane; k < CLSD; k += 32) acc = fmaf(qc[k], dc[k], acc);
        for (int off = 1; off < 32; off <<= 1) acc += __shfl_xor(acc, off);
    }

    if (lane == 0) out[q * BD + d] = sumtok + acc;
}

extern "C" void kernel_launch(void* const* d_in, const int* in_sizes, int n_in,
                              void* d_out, int out_size, void* d_ws, size_t ws_size,
                              hipStream_t stream) {
    (void)in_sizes; (void)n_in; (void)d_ws; (void)ws_size; (void)out_size;
    const float* qtok  = (const float*)d_in[0];   // [64,32,32]
    const float* dtok  = (const float*)d_in[1];   // [128,192,32]
    const float* qcls  = (const float*)d_in[2];   // [64,768]
    const float* dcls  = (const float*)d_in[3];   // [128,768]
    const int*   qid   = (const int*)d_in[4];     // [64,32]
    const int*   did   = (const int*)d_in[5];     // [128,192]
    const int*   qmask = (const int*)d_in[6];     // [64,32]
    float*       out   = (float*)d_out;           // [64,128]

    dim3 grid(BD, BQ / QPB);   // (doc, query-group); 8 wave32s per block
    dim3 block(32 * QPB);
    coil_score_kernel<<<grid, block, 0, stream>>>(qtok, dtok, qcls, dcls,
                                                  qid, did, qmask, out);
}
